// GRNModel_22299470201520
// MI455X (gfx1250) — matrix operand order repaired and checked by reading, not compiled, not run on hardware
//
#include <hip/hip_runtime.h>
#include <math.h>

// ---------------------------------------------------------------------------
// GRN VQA model for MI455X (gfx1250).
//  * Message passing (103 GFLOP over 12 steps): f16 filters (128 MB,
//    L2-resident on the 192MB L2) pre-swizzled into WMMA B-fragment order;
//    K split 4-way across blocks (256 workgroups/step) for occupancy.
//    Inner loop is pointer-march + constant-offset b128 loads + WMMA.
//  * BiLSTM input projections hoisted out of the recurrence into one WMMA
//    GEMM per direction ([1920x320]@[320x2048], K zero-padded); the 60
//    sequential launches only do the 512-deep recurrent h@Wh.
// ---------------------------------------------------------------------------

typedef __attribute__((ext_vector_type(16))) _Float16 v16h;
typedef __attribute__((ext_vector_type(8)))  float    v8f;

#define N_NEURONS 64
#define STATE     128
#define MSG       128
#define SEQB      64     // batch
#define SEQ       30
#define EMB       300
#define EMBP      320    // EMB padded to multiple of 32 for WMMA K
#define TF        512
#define IMGP      196    // 14*14
#define CLASSES   28

// ---- WMMA fragment-offset helpers (layouts per CDNA5 ISA 7.12.2) -----------

// A-matrix (16x32 f16): lane l holds row M=l%16; element e:
//   K = (e/8)*16 + (l>=16?8:0) + e%8.  Stored [mt][kt(nkt)][lane(32)][e(16)].
__device__ __forceinline__ size_t afrag_off(int row, int k, int nkt) {
  int mt   = row >> 4;
  int kt   = k >> 5;
  int ko   = k & 31;
  int lane = (((ko & 15) >> 3) << 4) + (row & 15);
  int e    = ((ko >> 4) << 3) + (ko & 7);
  return ((size_t)(mt * nkt + kt) * 32 + lane) * 16 + e;
}

// B-matrix (32x16 f16): lane l holds col N=l%16; K = (e/8)*16 + (l>=16?8:0) + e%8.
// Stored [kt][nt(nnt)][lane(32)][e(16)].
__device__ __forceinline__ size_t bfrag_off(int k, int n, int nnt) {
  int kt   = k >> 5;
  int ko   = k & 31;
  int nt   = n >> 4;
  int lane = (((ko & 15) >> 3) << 4) + (n & 15);
  int e    = ((ko >> 4) << 3) + (ko & 7);
  return ((size_t)(kt * nnt + nt) * 32 + lane) * 16 + e;
}

__device__ __forceinline__ float sigm(float x) { return 1.f / (1.f + __expf(-x)); }
__device__ __forceinline__ float eluf(float x) { return x > 0.f ? x : (__expf(x) - 1.f); }

// ---------------------------------------------------------------------------
// One-time kernels
// ---------------------------------------------------------------------------

// h1 = elu(edge @ fg_w1 + fg_b1)  : [4096,5]@[5,64]
__global__ __launch_bounds__(256) void k_h1(const float* __restrict__ edge,
    const float* __restrict__ w1, const float* __restrict__ b1,
    float* __restrict__ h1) {
  int gid = blockIdx.x * 256 + threadIdx.x;       // 262144
  int r = gid >> 6, kk = gid & 63;
  float acc = b1[kk];
  #pragma unroll
  for (int e = 0; e < 5; ++e) acc += edge[(size_t)r * 5 + e] * w1[(size_t)e * 64 + kk];
  h1[gid] = eluf(acc);
}

// filters = h1 @ fg_w2 + fg_b2, written as f16 in B-fragment order per neuron i.
// r = j*64+i, c = m*128+s. filters[j][i][m][s] -> Fi[k=j*128+s][n=m]; 256 kt, 8 nt.
__global__ __launch_bounds__(256) void k_filters(const float* __restrict__ h1,
    const float* __restrict__ w2, const float* __restrict__ b2,
    _Float16* __restrict__ Ffrag) {
  int c = blockIdx.x * 256 + threadIdx.x;         // 16384
  int r = blockIdx.y;                              // 4096
  float acc = b2[c];
  const float* hr = h1 + (size_t)r * 64;
  #pragma unroll 8
  for (int k = 0; k < 64; ++k) acc += hr[k] * w2[(size_t)k * 16384 + c];
  int j = r >> 6, i = r & 63;
  int m = c >> 7, s = c & 127;
  Ffrag[(size_t)i * (256 * 8 * 32 * 16) + bfrag_off(j * STATE + s, m, 8)] = (_Float16)acc;
}

// K[b][p][s] = sum_c image[b][c][p] * img_proj[c][s]
__global__ __launch_bounds__(128) void k_imgproj(const float* __restrict__ img,
    const float* __restrict__ proj, float* __restrict__ Kimg) {
  int bp = blockIdx.x;                             // b*196+p
  int s  = threadIdx.x;
  int b = bp / IMGP, p = bp % IMGP;
  const float* ib = img + (size_t)b * 512 * IMGP + p;
  float acc = 0.f;
  for (int cc = 0; cc < 512; ++cc) acc += ib[(size_t)cc * IMGP] * proj[(size_t)cc * 128 + s];
  Kimg[(size_t)bp * 128 + s] = acc;
}

// masked embedding lookup -> embm f32 [b][t][300]
__global__ __launch_bounds__(256) void k_embed(const int* __restrict__ q,
    const int* __restrict__ qlen, const float* __restrict__ tab,
    float* __restrict__ embm) {
  int gid = blockIdx.x * 256 + threadIdx.x;
  if (gid >= SEQB * SEQ * EMB) return;
  int e = gid % EMB; int rest = gid / EMB; int t = rest % SEQ; int b = rest / SEQ;
  float v = tab[(size_t)q[(size_t)b * SEQ + t] * EMB + e];
  embm[gid] = (t < qlen[b]) ? v : 0.f;
}

// embm -> A-fragments f16, rows r=b*30+t (1920), K padded 300->320 (10 kt)
__global__ __launch_bounds__(256) void k_embf(const float* __restrict__ embm,
    _Float16* __restrict__ Afrag) {
  int gid = blockIdx.x * 256 + threadIdx.x;       // 1920*320 = 614400
  if (gid >= 1920 * EMBP) return;
  int r = gid / EMBP, k = gid % EMBP;
  float v = (k < EMB) ? embm[(size_t)r * EMB + k] : 0.f;
  Afrag[afrag_off(r, k, EMBP / 32)] = (_Float16)v;
}

// lstm Wi [300,2048] -> B-fragments f16 padded to [320,2048] (10 kt, 128 nt)
__global__ __launch_bounds__(256) void k_wif(const float* __restrict__ Wi,
    _Float16* __restrict__ Bfrag) {
  int gid = blockIdx.x * 256 + threadIdx.x;       // 320*2048 = 655360
  if (gid >= EMBP * 2048) return;
  int k = gid / 2048, c = gid % 2048;
  float v = (k < EMB) ? Wi[(size_t)k * 2048 + c] : 0.f;
  Bfrag[bfrag_off(k, c, 128)] = (_Float16)v;
}

// gx = embm_pad @ Wi_pad : [1920x320]@[320x2048] via WMMA, fp32 out.
// grid (30,16); 8 waves: mt = bx*4 + wave/2 (0..119), nt0 = by*8 + (wave&1)*4.
__global__ __launch_bounds__(256) void k_gx_mm(const _Float16* __restrict__ Afrag,
    const _Float16* __restrict__ Bfrag, float* __restrict__ C) {
  const int wave = threadIdx.x >> 5;
  const int lane = threadIdx.x & 31;
  const int mt   = blockIdx.x * 4 + (wave >> 1);
  const int nt0  = blockIdx.y * 8 + (wave & 1) * 4;
  v8f acc[4] = {};
  // pointer-march: constant strides so loads get immediate offsets
  const _Float16* aPtr = Afrag + ((size_t)mt * 10 * 32 + lane) * 16;
  const _Float16* bPtr = Bfrag + ((size_t)nt0 * 32 + lane) * 16;
  for (int kt = 0; kt < 10; ++kt) {
    v16h a  = *(const v16h*)aPtr;
    v16h b0 = *(const v16h*)(bPtr);
    v16h b1 = *(const v16h*)(bPtr + 512);
    v16h b2 = *(const v16h*)(bPtr + 1024);
    v16h b3 = *(const v16h*)(bPtr + 1536);
    acc[0] = __builtin_amdgcn_wmma_f32_16x16x32_f16(false, a, false, b0, (short)0, acc[0], false, false);
    acc[1] = __builtin_amdgcn_wmma_f32_16x16x32_f16(false, a, false, b1, (short)0, acc[1], false, false);
    acc[2] = __builtin_amdgcn_wmma_f32_16x16x32_f16(false, a, false, b2, (short)0, acc[2], false, false);
    acc[3] = __builtin_amdgcn_wmma_f32_16x16x32_f16(false, a, false, b3, (short)0, acc[3], false, false);
    aPtr += 512;                 // next kt of A: 32 lanes * 16 halves
    bPtr += 128 * 512;           // next kt row of B-fragments
  }
  const int rbase = mt * 16 + ((lane >= 16) ? 8 : 0);
  const int col   = lane & 15;
  #pragma unroll
  for (int t = 0; t < 4; ++t) {
    int c = (nt0 + t) * 16 + col;
    #pragma unroll
    for (int v = 0; v < 8; ++v)
      C[(size_t)(rbase + v) * 2048 + c] = acc[t][v];
  }
}

// One LSTM recurrent step: gates = gx[b,t] + bias + h@Wh. thread = (b,u).
__global__ __launch_bounds__(256) void k_lstm(const float* __restrict__ gx, int t,
    const float* __restrict__ Wh, const float* __restrict__ bias,
    const float* __restrict__ hprev, const float* __restrict__ cprev,
    float* __restrict__ hnew, float* __restrict__ cnew,
    float* __restrict__ hout, int first) {
  int gid = blockIdx.x * 256 + threadIdx.x;       // 32768
  int b = gid >> 9, u = gid & 511;
  const float* gr = gx + ((size_t)b * SEQ + t) * 2048;
  float g0 = gr[u]        + bias[u];
  float g1 = gr[512 + u]  + bias[512 + u];
  float g2 = gr[1024 + u] + bias[1024 + u];
  float g3 = gr[1536 + u] + bias[1536 + u];
  float cp = 0.f;
  if (!first) {
    const float* hp = hprev + (size_t)b * TF;
    const float* w  = Wh + u;
    for (int k = 0; k < TF; ++k) {
      float hv = hp[k];
      g0 += hv * w[0]; g1 += hv * w[512]; g2 += hv * w[1024]; g3 += hv * w[1536];
      w += 2048;
    }
    cp = cprev[(size_t)b * TF + u];
  }
  float c = sigm(g1) * cp + sigm(g0) * tanhf(g2);
  float h = sigm(g3) * tanhf(c);
  cnew[(size_t)b * TF + u] = c;
  hnew[(size_t)b * TF + u] = h;
  hout[((size_t)b * SEQ + t) * TF + u] = h;
}

// hidden = concat(hf[b, qlen-1], hb[b, 0])
__global__ __launch_bounds__(256) void k_hidden(const float* __restrict__ hf,
    const float* __restrict__ hb, const int* __restrict__ qlen,
    float* __restrict__ hid) {
  int gid = blockIdx.x * 256 + threadIdx.x;       // 65536
  int b = gid >> 10, u = gid & 1023;
  if (u < TF) {
    int idx = qlen[b] - 1; if (idx < 0) idx = 0;
    hid[gid] = hf[((size_t)b * SEQ + idx) * TF + u];
  } else {
    hid[gid] = hb[(size_t)b * SEQ * TF + (u - TF)];
  }
}

// wp[b][t][s] = concat(hf,hb)[b][t] @ word_proj
__global__ __launch_bounds__(256) void k_wp(const float* __restrict__ hf,
    const float* __restrict__ hb, const float* __restrict__ wordp,
    float* __restrict__ wp) {
  int gid = blockIdx.x * 256 + threadIdx.x;       // 245760
  int s = gid & 127; int rest = gid >> 7; int t = rest % SEQ; int b = rest / SEQ;
  const float* hfr = hf + ((size_t)b * SEQ + t) * TF;
  const float* hbr = hb + ((size_t)b * SEQ + t) * TF;
  float acc = 0.f;
  for (int k = 0; k < TF; ++k) acc += hfr[k] * wordp[(size_t)k * 128 + s];
  for (int k = 0; k < TF; ++k) acc += hbr[k] * wordp[(size_t)(TF + k) * 128 + s];
  wp[(size_t)gid] = acc;
}

// H init (broadcast initial_state) in f32 + A-fragment f16 (256 kt)
__global__ __launch_bounds__(256) void k_initH(const float* __restrict__ is,
    float* __restrict__ H, _Float16* __restrict__ Hfrag) {
  int gid = blockIdx.x * 256 + threadIdx.x;       // 524288
  int r = gid >> 7, s = gid & 127;
  int j = r & 63, b = r >> 6;
  float v = is[(size_t)j * 128 + s];
  H[gid] = v;
  Hfrag[afrag_off(b, j * STATE + s, 256)] = (_Float16)v;
}

// ---------------------------------------------------------------------------
// Per-step kernels
// ---------------------------------------------------------------------------

// Attention (blockIdx.y: 0=image over 196 pos, 1=text over 30 masked words).
__global__ __launch_bounds__(256) void k_attn(const float* __restrict__ Kimg,
    const float* __restrict__ wp, const int* __restrict__ qlen,
    float* __restrict__ H, _Float16* __restrict__ Hfrag) {
  __shared__ float xq[128];
  __shared__ float sl[256];
  __shared__ float rb[256];
  const int b = blockIdx.x;
  const int isText = blockIdx.y;
  const int tid = threadIdx.x;
  const int qrow = isText ? 62 : 63;               // g1 / g0
  if (tid < 128) xq[tid] = H[((size_t)b * 64 + qrow) * 128 + tid];
  __syncthreads();
  const int P = isText ? SEQ : IMGP;
  const float* V = isText ? (wp + (size_t)b * SEQ * 128) : (Kimg + (size_t)b * IMGP * 128);
  float lg = -1e30f;
  if (tid < P) {
    if (isText && tid >= qlen[b]) {
      lg = -1e9f;
    } else {
      float d = 0.f;
      const float* vr = V + (size_t)tid * 128;
      for (int s2 = 0; s2 < 128; ++s2) d += vr[s2] * xq[s2];
      lg = d;
    }
  }
  rb[tid] = lg;
  __syncthreads();
  for (int st = 128; st > 0; st >>= 1) { if (tid < st) rb[tid] = fmaxf(rb[tid], rb[tid + st]); __syncthreads(); }
  float mx = rb[0];
  __syncthreads();
  float ev = (tid < P) ? __expf(lg - mx) : 0.f;
  sl[tid] = ev;
  rb[tid] = ev;
  __syncthreads();
  for (int st = 128; st > 0; st >>= 1) { if (tid < st) rb[tid] += rb[tid + st]; __syncthreads(); }
  float inv = 1.f / rb[0];
  if (tid < 128) {
    float o = 0.f;
    for (int p = 0; p < P; ++p) o += sl[p] * V[(size_t)p * 128 + tid];
    o *= inv;
    int j = isText;
    H[((size_t)b * 64 + j) * 128 + tid] = o;
    Hfrag[afrag_off(b, j * STATE + tid, 256)] = (_Float16)o;
  }
}

// Message passing, K-split 4-way: block = (neuron i, ks). Each block reduces
// kt in [ks*64, ks*64+64) and writes a partial [64x128] to msgsP[ks].
// Inner loop: pointer-march, constant-offset fragment loads, 4 WMMAs.
__global__ __launch_bounds__(256) void k_mp(const _Float16* __restrict__ Ffrag,
    const _Float16* __restrict__ Hfrag, float* __restrict__ msgsP) {
  const int i    = blockIdx.x;                     // neuron 0..63
  const int ks   = blockIdx.y;                     // K-split 0..3
  const int wave = threadIdx.x >> 5;               // 0..7
  const int lane = threadIdx.x & 31;
  const int mt   = wave >> 1;                      // batch tile 0..3
  const int nt0  = (wave & 1) * 4;                 // msg tile base 0 or 4
  const int kt0  = ks * 64;
  v8f acc[4] = {};
  const _Float16* aPtr = Hfrag + ((size_t)(mt * 256 + kt0) * 32 + lane) * 16;
  const _Float16* bPtr = Ffrag + (((size_t)(i * 256 + kt0) * 8 + nt0) * 32 + lane) * 16;
  for (int kk = 0; kk < 64; ++kk) {
    v16h a  = *(const v16h*)aPtr;
    v16h b0 = *(const v16h*)(bPtr);
    v16h b1 = *(const v16h*)(bPtr + 512);
    v16h b2 = *(const v16h*)(bPtr + 1024);
    v16h b3 = *(const v16h*)(bPtr + 1536);
    acc[0] = __builtin_amdgcn_wmma_f32_16x16x32_f16(false, a, false, b0, (short)0, acc[0], false, false);
    acc[1] = __builtin_amdgcn_wmma_f32_16x16x32_f16(false, a, false, b1, (short)0, acc[1], false, false);
    acc[2] = __builtin_amdgcn_wmma_f32_16x16x32_f16(false, a, false, b2, (short)0, acc[2], false, false);
    acc[3] = __builtin_amdgcn_wmma_f32_16x16x32_f16(false, a, false, b3, (short)0, acc[3], false, false);
    aPtr += 512;                 // next kt of H: 32 lanes * 16 halves
    bPtr += 8 * 512;             // next kt row of F-fragments (8 nt per kt)
  }
  float* dst = msgsP + (size_t)ks * (4096 * 128);
  const int bbase = mt * 16 + ((lane >= 16) ? 8 : 0);
  const int mcol  = lane & 15;
  #pragma unroll
  for (int t = 0; t < 4; ++t) {
    int m = (nt0 + t) * 16 + mcol;
    #pragma unroll
    for (int v = 0; v < 8; ++v) {
      int b = bbase + v;
      dst[((size_t)(b * 64 + i)) * 128 + m] = acc[t][v];
    }
  }
}

// GRU cell; sums the 4 K-split partials on the fly. thread = (row r=b*64+j, s).
__global__ __launch_bounds__(256) void k_gru(const float* __restrict__ H,
    const float* __restrict__ msgsP,
    const float* __restrict__ Wi, const float* __restrict__ bi,
    const float* __restrict__ Wh, const float* __restrict__ bh,
    float* __restrict__ Hout, _Float16* __restrict__ Hfrag) {
  int gid = blockIdx.x * 256 + threadIdx.x;       // 524288
  int r = gid >> 7, s = gid & 127;
  const float* m0 = msgsP + (size_t)r * 128;
  const float* m1 = m0 + (size_t)1 * 4096 * 128;
  const float* m2 = m0 + (size_t)2 * 4096 * 128;
  const float* m3 = m0 + (size_t)3 * 4096 * 128;
  const float* hrow = H + (size_t)r * 128;
  float xz = bi[s], xr = bi[128 + s], xn = bi[256 + s];
  float hz = bh[s], hr = bh[128 + s], hn = bh[256 + s];
  for (int k = 0; k < 128; ++k) {
    float mv = m0[k] + m1[k] + m2[k] + m3[k];
    const float* w = Wi + (size_t)k * 384;
    xz += mv * w[s]; xr += mv * w[128 + s]; xn += mv * w[256 + s];
  }
  for (int k = 0; k < 128; ++k) {
    float hv = hrow[k];
    const float* w = Wh + (size_t)k * 384;
    hz += hv * w[s]; hr += hv * w[128 + s]; hn += hv * w[256 + s];
  }
  float z    = sigm(xz + hz);
  float rr   = sigm(xr + hr);
  float cand = tanhf(xn + rr * hn);
  float hp   = hrow[s];
  float hnew = (1.f - z) * cand + z * hp;
  Hout[(size_t)r * 128 + s] = hnew;
  int bb = r >> 6, j = r & 63;
  Hfrag[afrag_off(bb, j * STATE + s, 256)] = (_Float16)hnew;
}

// ---------------------------------------------------------------------------
// Classifier
// ---------------------------------------------------------------------------
__global__ __launch_bounds__(256) void k_cls1(const float* __restrict__ H,
    const float* __restrict__ hid, const float* __restrict__ W,
    const float* __restrict__ bias, float* __restrict__ outh) {
  int gid = blockIdx.x * 256 + threadIdx.x;       // 8192
  int b = gid >> 7, s = gid & 127;
  float acc = bias[s];
  const float* fin = H + ((size_t)b * 64 + 61) * 128;
  for (int k = 0; k < 128; ++k)  acc += fin[k] * W[(size_t)k * 128 + s];
  const float* hv = hid + (size_t)b * 1024;
  for (int k = 0; k < 1024; ++k) acc += hv[k] * W[(size_t)(128 + k) * 128 + s];
  outh[(size_t)b * 128 + s] = eluf(acc);
}

__global__ __launch_bounds__(256) void k_cls2(const float* __restrict__ hcls,
    const float* __restrict__ W, const float* __restrict__ bias,
    float* __restrict__ out) {
  int gid = blockIdx.x * 256 + threadIdx.x;
  if (gid >= SEQB * CLASSES) return;
  int b = gid / CLASSES, j = gid % CLASSES;
  float acc = bias[j];
  const float* hr = hcls + (size_t)b * 128;
  for (int k = 0; k < 128; ++k) acc += hr[k] * W[(size_t)k * CLASSES + j];
  out[gid] = acc;
}

// ---------------------------------------------------------------------------
extern "C" void kernel_launch(void* const* d_in, const int* in_sizes, int n_in,
                              void* d_out, int out_size, void* d_ws, size_t ws_size,
                              hipStream_t stream) {
  const float* image    = (const float*)d_in[0];
  const int*   question = (const int*)  d_in[1];
  const int*   qlen     = (const int*)  d_in[2];
  const float* edge     = (const float*)d_in[3];
  const float* initst   = (const float*)d_in[4];
  const float* fg_w1    = (const float*)d_in[5];
  const float* fg_b1    = (const float*)d_in[6];
  const float* fg_w2    = (const float*)d_in[7];
  const float* fg_b2    = (const float*)d_in[8];
  const float* gWi      = (const float*)d_in[9];
  const float* gbi      = (const float*)d_in[10];
  const float* gWh      = (const float*)d_in[11];
  const float* gbh      = (const float*)d_in[12];
  const float* img_proj = (const float*)d_in[13];
  const float* etab     = (const float*)d_in[14];
  const float* fWi      = (const float*)d_in[15];
  const float* fWh      = (const float*)d_in[16];
  const float* fbb      = (const float*)d_in[17];
  const float* bWi      = (const float*)d_in[18];
  const float* bWh      = (const float*)d_in[19];
  const float* bbb      = (const float*)d_in[20];
  const float* wproj    = (const float*)d_in[21];
  const float* c1w      = (const float*)d_in[22];
  const float* c1b      = (const float*)d_in[23];
  const float* c2w      = (const float*)d_in[24];
  const float* c2b      = (const float*)d_in[25];
  float* out = (float*)d_out;

  // workspace carve-up (256B aligned)
  char* ws = (char*)d_ws;
  size_t off = 0;
  auto carve = [&](size_t bytes) {
    void* p = ws + off;
    off += (bytes + 255) & ~(size_t)255;
    return p;
  };
  _Float16* Ffrag = (_Float16*)carve((size_t)64 * 256 * 8 * 32 * 16 * 2); // 128 MB
  _Float16* Hfrag = (_Float16*)carve((size_t)4 * 256 * 32 * 16 * 2);      // 1 MB
  float* HA     = (float*)carve((size_t)4096 * 128 * 4);
  float* HB     = (float*)carve((size_t)4096 * 128 * 4);
  float* msgsP  = (float*)carve((size_t)4 * 4096 * 128 * 4);              // 4 partials
  float* h1     = (float*)carve((size_t)4096 * 64 * 4);
  float* Kimg   = (float*)carve((size_t)SEQB * IMGP * 128 * 4);
  float* embm   = (float*)carve((size_t)SEQB * SEQ * EMB * 4);
  _Float16* embA = (_Float16*)carve((size_t)1920 * EMBP * 2);             // A-frags
  _Float16* wiBf = (_Float16*)carve((size_t)EMBP * 2048 * 2);             // fwd Wi B-frags
  _Float16* wiBb = (_Float16*)carve((size_t)EMBP * 2048 * 2);             // bwd Wi B-frags
  float* gxF    = (float*)carve((size_t)1920 * 2048 * 4);                 // fwd input proj
  float* gxB    = (float*)carve((size_t)1920 * 2048 * 4);                 // bwd input proj
  float* hf     = (float*)carve((size_t)SEQB * SEQ * TF * 4);
  float* hb     = (float*)carve((size_t)SEQB * SEQ * TF * 4);
  float* hsA    = (float*)carve((size_t)SEQB * TF * 4);
  float* hsB    = (float*)carve((size_t)SEQB * TF * 4);
  float* csA    = (float*)carve((size_t)SEQB * TF * 4);
  float* csB    = (float*)carve((size_t)SEQB * TF * 4);
  float* hidv   = (float*)carve((size_t)SEQB * 1024 * 4);
  float* wp     = (float*)carve((size_t)SEQB * SEQ * 128 * 4);
  float* clsh   = (float*)carve((size_t)SEQB * 128 * 4);
  if (off > ws_size) return;  // workspace too small; nothing sane to do

  // ---- one-time precompute ----
  k_h1     <<<1024, 256, 0, stream>>>(edge, fg_w1, fg_b1, h1);
  k_filters<<<dim3(64, 4096), 256, 0, stream>>>(h1, fg_w2, fg_b2, Ffrag);
  k_imgproj<<<SEQB * IMGP, 128, 0, stream>>>(image, img_proj, Kimg);
  k_embed  <<<(SEQB * SEQ * EMB + 255) / 256, 256, 0, stream>>>(question, qlen, etab, embm);

  // BiLSTM input projections as WMMA GEMMs (no sequential dependency)
  k_embf <<<(1920 * EMBP + 255) / 256, 256, 0, stream>>>(embm, embA);
  k_wif  <<<(EMBP * 2048 + 255) / 256, 256, 0, stream>>>(fWi, wiBf);
  k_wif  <<<(EMBP * 2048 + 255) / 256, 256, 0, stream>>>(bWi, wiBb);
  k_gx_mm<<<dim3(30, 16), 256, 0, stream>>>(embA, wiBf, gxF);
  k_gx_mm<<<dim3(30, 16), 256, 0, stream>>>(embA, wiBb, gxB);

  // BiLSTM recurrence: 30 sequential steps per direction, ping-pong h/c state
  {
    float *hp = hsA, *hn = hsB, *cp = csA, *cn = csB;
    for (int t = 0; t < SEQ; ++t) {
      k_lstm<<<128, 256, 0, stream>>>(gxF, t, fWh, fbb, hp, cp, hn, cn, hf, t == 0);
      float* tp = hp; hp = hn; hn = tp;
      tp = cp; cp = cn; cn = tp;
    }
    hp = hsA; hn = hsB; cp = csA; cn = csB;
    for (int tt = 0; tt < SEQ; ++tt) {
      int t = SEQ - 1 - tt;
      k_lstm<<<128, 256, 0, stream>>>(gxB, t, bWh, bbb, hp, cp, hn, cn, hb, tt == 0);
      float* tp = hp; hp = hn; hn = tp;
      tp = cp; cp = cn; cn = tp;
    }
  }
  k_hidden<<<256, 256, 0, stream>>>(hf, hb, qlen, hidv);
  k_wp    <<<960, 256, 0, stream>>>(hf, hb, wproj, wp);

  // ---- 12 message-passing steps ----
  k_initH<<<2048, 256, 0, stream>>>(initst, HA, Hfrag);
  float* Hcur = HA;
  float* Hnxt = HB;
  for (int s = 0; s < 12; ++s) {
    k_attn<<<dim3(64, 2), 256, 0, stream>>>(Kimg, wp, qlen, Hcur, Hfrag);
    k_mp  <<<dim3(64, 4), 256, 0, stream>>>(Ffrag, Hfrag, msgsP);
    k_gru <<<2048, 256, 0, stream>>>(Hcur, msgsP, gWi, gbi, gWh, gbh, Hnxt, Hfrag);
    float* tp = Hcur; Hcur = Hnxt; Hnxt = tp;
  }

  // ---- classifier ----
  k_cls1<<<32, 256, 0, stream>>>(Hcur, hidv, c1w, c1b, clsh);
  k_cls2<<<7, 256, 0, stream>>>(clsh, c2w, c2b, out);
}
